// MultiHeadAttention_19662360281546
// MI455X (gfx1250) — matrix operand order repaired and checked by reading, not compiled, not run on hardware
//
#include <hip/hip_runtime.h>

typedef __attribute__((ext_vector_type(16))) __bf16 v16bf;
typedef __attribute__((ext_vector_type(8)))  __bf16 v8bf;
typedef __attribute__((ext_vector_type(8)))  float  v8f;

#define NPIX 4096   // 64*64
#define CCH  128
#define DH   32     // head dim
#define NHEAD 4
#define QSCALE 0.17677669529663687f  // 32^-0.5

__device__ __forceinline__ v8f wmma_bf16(v16bf a, v16bf b, v8f c) {
  return __builtin_amdgcn_wmma_f32_16x16x32_bf16(false, a, false, b, (short)0, c,
                                                 false, false);
}

// Async global->LDS copy of one 16B chunk (per-lane), CDNA5 ASYNCcnt path.
__device__ __forceinline__ void async_copy_b128(const void* gptr, void* lptr) {
  unsigned long long ga = (unsigned long long)gptr;
  unsigned ldsoff = (unsigned)(unsigned long long)lptr;  // low 32b = LDS offset
  asm volatile("global_load_async_to_lds_b128 %0, %1, off"
               :: "v"(ldsoff), "v"(ga) : "memory");
}
__device__ __forceinline__ void wait_async() {
  asm volatile("s_wait_asynccnt 0" ::: "memory");
}

// ---------------------------------------------------------------------------
// Kernel 0: one-time prep — convert weights fp32->bf16, zero GN accumulators.
// grid = 192 x 256 (exactly 49152 = 384*128 threads)
// ---------------------------------------------------------------------------
__global__ void prep_kernel(const float* __restrict__ wqkv,
                            const float* __restrict__ wout,
                            __bf16* __restrict__ wqkv_bf,
                            __bf16* __restrict__ wout_bf,
                            float* __restrict__ gsum) {
  const int i = blockIdx.x * 256 + threadIdx.x;
  wqkv_bf[i] = (__bf16)wqkv[i];
  if (i < CCH * CCH) wout_bf[i] = (__bf16)wout[i];
  if (i < 32) gsum[i] = 0.0f;
}

// ---------------------------------------------------------------------------
// Kernel 1: QKV projection.  qkv[o,p] = sum_c w_qkv[o,c] * x[c,p]
// Writes Q (pre-scaled by 1/sqrt(d)), K as bf16 [bh][n][d]; V^T bf16 [bh][d][n].
// Block = 256 thr (8 waves) -> 128 pixels; grid = (32 ptiles, 4 batches).
// ---------------------------------------------------------------------------
__global__ void __launch_bounds__(256) qkv_kernel(
    const float* __restrict__ x, const __bf16* __restrict__ w_qkv,
    __bf16* __restrict__ Qb, __bf16* __restrict__ Kb, __bf16* __restrict__ Vt) {
  __shared__ alignas(32) __bf16 Xt[128][CCH];  // x tile transposed: [p_local][c]
  const int tid = threadIdx.x;
  const int b = blockIdx.y;
  const int p0 = blockIdx.x * 128;
  const float* xb = x + (size_t)b * CCH * NPIX;

  // Stage + transpose + convert x tile: each thread loads 64 contiguous floats
  {
    int c = tid >> 1;
    int pseg = (tid & 1) * 64;
    const float4* src = (const float4*)(xb + (size_t)c * NPIX + p0 + pseg);
#pragma unroll
    for (int j = 0; j < 16; ++j) {
      float4 v = src[j];
      int pl = pseg + j * 4;
      Xt[pl + 0][c] = (__bf16)v.x;
      Xt[pl + 1][c] = (__bf16)v.y;
      Xt[pl + 2][c] = (__bf16)v.z;
      Xt[pl + 3][c] = (__bf16)v.w;
    }
  }
  __syncthreads();

  const int lane = tid & 31;
  const int wv = tid >> 5;
  const int m = lane & 15;
  const int half = lane >> 4;
  const int plocal = wv * 16;

  for (int t = 0; t < 24; ++t) {  // 24 o-tiles of 16 over 384 outputs
    v8f acc = {};
#pragma unroll
    for (int cc = 0; cc < 4; ++cc) {
      const int cbase = cc * 32;
      // A fragment: pre-converted bf16 weight row, one contiguous 32B load
      const __bf16* wr = w_qkv + (size_t)(t * 16 + m) * CCH + cbase;
      v16bf af = *(const v16bf*)(wr + half * 16);
      // B fragment: contiguous 32B from transposed LDS tile
      v16bf bf = *(const v16bf*)&Xt[plocal + m][cbase + half * 16];
      acc = wmma_bf16(af, bf, acc);
    }
    // Scatter C tile: rows = out channels, cols = pixels (col = lane&15)
    const int o0 = t * 16;
    const int third = o0 >> 7;        // 0=q 1=k 2=v
    const int oc0 = o0 & 127;
    const int head = oc0 >> 5;
    const int dbase = oc0 & 31;       // 0 or 16
    const int p = p0 + plocal + m;    // this lane's pixel
    const size_t bh = (size_t)(b * NHEAD + head);
    if (third < 2) {
      const float sc = (third == 0) ? QSCALE : 1.0f;
      __bf16* dst = ((third == 0) ? Qb : Kb) +
                    (bh * NPIX + p) * DH + dbase + half * 8;
      v8bf pk;
#pragma unroll
      for (int r = 0; r < 8; ++r) pk[r] = (__bf16)(acc[r] * sc);
      *(v8bf*)dst = pk;  // 8 consecutive d values -> one 16B store
    } else {
#pragma unroll
      for (int r = 0; r < 8; ++r) {
        int d = dbase + half * 8 + r;
        Vt[(bh * DH + d) * NPIX + p] = (__bf16)acc[r];
      }
    }
  }
}

// ---------------------------------------------------------------------------
// Kernel 2: Flash attention per (b,head).  Wave owns 16 query rows, block 128.
// grid = (32 qtiles, 16 bh).  KV tiles of 32 staged to LDS once per block via
// async global->LDS (ASYNCcnt); all fragment reads served from LDS.
// Output O[b][pixel][head*32+d] bf16 (ready for out-projection B-fragments).
// ---------------------------------------------------------------------------
__global__ void __launch_bounds__(256) attn_kernel(
    const __bf16* __restrict__ Qb, const __bf16* __restrict__ Kb,
    const __bf16* __restrict__ Vt, __bf16* __restrict__ O) {
  __shared__ alignas(32) __bf16 Klds[32][DH];     // [key_local][d]   2 KB
  __shared__ alignas(32) __bf16 Vlds[32][32];     // [d][key_local]   2 KB
  __shared__ alignas(32) __bf16 Plds[8][16][32];  // per-wave P tile [q][key]
  const int tid = threadIdx.x;
  const int lane = tid & 31;
  const int wv = tid >> 5;
  const int bh = blockIdx.y;
  const int q0 = blockIdx.x * 128 + wv * 16;
  const int m = lane & 15;
  const int half = lane >> 4;

  // Q A-fragment (row q0+m, contiguous 16 d per half)
  const __bf16* qp = Qb + ((size_t)bh * NPIX + q0 + m) * DH;
  v16bf qf = *(const v16bf*)(qp + half * 16);

  v8f acc0 = {}, acc1 = {};  // out accumulator, d=0..15 and d=16..31
  float mrow[8], lrow[8];
#pragma unroll
  for (int r = 0; r < 8; ++r) { mrow[r] = -1e30f; lrow[r] = 0.0f; }

  const __bf16* kp = Kb + (size_t)bh * NPIX * DH;
  const __bf16* vp = Vt + (size_t)bh * DH * NPIX;

  // Per-thread async staging assignment: one 16B chunk each (256*16B = 4KB)
  const int srow = (tid & 127) >> 2;  // 0..31
  const int ssub = tid & 3;           // 16B chunk within a 64B row

  for (int kv = 0; kv < NPIX; kv += 32) {
    // ---- stage K/V tile to LDS (async, ASYNCcnt) ----
    if (tid < 128) {
      async_copy_b128(kp + (size_t)(kv + srow) * DH + ssub * 8,
                      &Klds[srow][ssub * 8]);
    } else {
      async_copy_b128(vp + (size_t)srow * NPIX + kv + ssub * 8,
                      &Vlds[srow][ssub * 8]);
    }
    wait_async();
    __syncthreads();

    // ---- S = Q.K^T : B-fragments straight from LDS ----
    v16bf kf0 = *(const v16bf*)&Klds[m][half * 16];
    v16bf kf1 = *(const v16bf*)&Klds[16 + m][half * 16];
    v8f s0 = {}, s1 = {};
    s0 = wmma_bf16(qf, kf0, s0);   // keys kv..kv+15
    s1 = wmma_bf16(qf, kf1, s1);   // keys kv+16..kv+31

    // ---- online softmax over this 32-key tile ----
#pragma unroll
    for (int r = 0; r < 8; ++r) {
      float tm = fmaxf(s0[r], s1[r]);
      tm = fmaxf(tm, __shfl_xor(tm, 1, 32));
      tm = fmaxf(tm, __shfl_xor(tm, 2, 32));
      tm = fmaxf(tm, __shfl_xor(tm, 4, 32));
      tm = fmaxf(tm, __shfl_xor(tm, 8, 32));
      float mnew = fmaxf(mrow[r], tm);
      float corr = __expf(mrow[r] - mnew);
      mrow[r] = mnew;
      float p0v = __expf(s0[r] - mnew);
      float p1v = __expf(s1[r] - mnew);
      float ts = p0v + p1v;
      ts += __shfl_xor(ts, 1, 32);
      ts += __shfl_xor(ts, 2, 32);
      ts += __shfl_xor(ts, 4, 32);
      ts += __shfl_xor(ts, 8, 32);
      lrow[r] = lrow[r] * corr + ts;
      acc0[r] *= corr;
      acc1[r] *= corr;
      const int row = r + half * 8;
      Plds[wv][row][m] = (__bf16)p0v;       // C-layout -> LDS [q][key]
      Plds[wv][row][16 + m] = (__bf16)p1v;
    }

    // ---- P A-fragment (per-wave LDS region; same-wave DS ops are in-order) --
    v16bf pf = *(const v16bf*)&Plds[wv][m][half * 16];

    // ---- PV : V B-fragments from LDS ----
    v16bf vf0 = *(const v16bf*)&Vlds[m][half * 16];
    v16bf vf1 = *(const v16bf*)&Vlds[16 + m][half * 16];
    acc0 = wmma_bf16(pf, vf0, acc0);
    acc1 = wmma_bf16(pf, vf1, acc1);

    __syncthreads();  // protect Klds/Vlds before next tile's async stores
  }

  // Epilogue: normalize by l and store O[b][pixel][head*32+d]
  const int b = bh >> 2;
  const int head = bh & 3;
#pragma unroll
  for (int r = 0; r < 8; ++r) {
    const int row = r + half * 8;
    const float inv = 1.0f / lrow[r];
    const size_t base = ((size_t)b * NPIX + q0 + row) * CCH + head * DH;
    O[base + m]      = (__bf16)(acc0[r] * inv);
    O[base + 16 + m] = (__bf16)(acc1[r] * inv);
  }
}

// ---------------------------------------------------------------------------
// Kernel 3: out-projection y[b][c][p] = W_out @ O + b_out, plus per-(b,group)
// sum / sumsq accumulation for GroupNorm.  grid = (32 ptiles, 4 batches).
// ---------------------------------------------------------------------------
__global__ void __launch_bounds__(256) outproj_kernel(
    const __bf16* __restrict__ O, const __bf16* __restrict__ w_out,
    const float* __restrict__ b_out, float* __restrict__ y,
    float* __restrict__ gsum, float* __restrict__ gss) {
  __shared__ float lsum[4], lss[4];
  const int tid = threadIdx.x;
  if (tid < 4) { lsum[tid] = 0.0f; lss[tid] = 0.0f; }
  __syncthreads();

  const int b = blockIdx.y;
  const int p0 = blockIdx.x * 128;
  const int lane = tid & 31;
  const int wv = tid >> 5;
  const int m = lane & 15;
  const int half = lane >> 4;
  const int p = p0 + wv * 16 + m;  // this lane's pixel column
  const __bf16* Ob = O + ((size_t)b * NPIX + p0 + wv * 16) * CCH;

  float psum = 0.0f, pss = 0.0f;
  for (int t = 0; t < 8; ++t) {
    v8f acc = {};
#pragma unroll
    for (int cc = 0; cc < 4; ++cc) {
      const int cbase = cc * 32;
      const __bf16* wr = w_out + (size_t)(t * 16 + m) * CCH + cbase;
      v16bf af = *(const v16bf*)(wr + half * 16);
      v16bf bf = *(const v16bf*)(Ob + (size_t)m * CCH + cbase + half * 16);
      acc = wmma_bf16(af, bf, acc);
    }
#pragma unroll
    for (int r = 0; r < 8; ++r) {
      const int o2 = t * 16 + half * 8 + r;
      const float v = acc[r] + b_out[o2];
      y[((size_t)b * CCH + o2) * NPIX + p] = v;
      psum += v;
      pss += v * v;
    }
    if (t & 1) {  // finished a 32-channel group (tiles 2g, 2g+1)
      atomicAdd(&lsum[t >> 1], psum);
      atomicAdd(&lss[t >> 1], pss);
      psum = 0.0f;
      pss = 0.0f;
    }
  }
  __syncthreads();
  if (tid < 4) {
    atomicAdd(&gsum[b * 4 + tid], lsum[tid]);
    atomicAdd(&gss[b * 4 + tid], lss[tid]);
  }
}

// ---------------------------------------------------------------------------
// Kernel 4: GroupNorm(4) + affine + residual, float4-vectorized elementwise
// ---------------------------------------------------------------------------
__global__ void __launch_bounds__(256) gn_kernel(
    const float* __restrict__ y, const float* __restrict__ x,
    const float* __restrict__ gamma, const float* __restrict__ beta,
    const float* __restrict__ gsum, const float* __restrict__ gss,
    float* __restrict__ out) {
  const size_t i = ((size_t)blockIdx.x * blockDim.x + threadIdx.x) * 4;
  const int c = (int)((i >> 12) & 127);  // /4096 % 128
  const int b = (int)(i >> 19);          // /(128*4096)
  const int g = b * 4 + (c >> 5);
  const float invN = 1.0f / (32.0f * 4096.0f);
  const float mean = gsum[g] * invN;
  const float var = gss[g] * invN - mean * mean;
  const float rinv = rsqrtf(var + 1e-5f);
  const float ga = gamma[c], be = beta[c];
  const float4 yv = *(const float4*)(y + i);
  const float4 xv = *(const float4*)(x + i);
  float4 o;
  o.x = (yv.x - mean) * rinv * ga + be + xv.x;
  o.y = (yv.y - mean) * rinv * ga + be + xv.y;
  o.z = (yv.z - mean) * rinv * ga + be + xv.z;
  o.w = (yv.w - mean) * rinv * ga + be + xv.w;
  *(float4*)(out + i) = o;
}

// ---------------------------------------------------------------------------
extern "C" void kernel_launch(void* const* d_in, const int* in_sizes, int n_in,
                              void* d_out, int out_size, void* d_ws,
                              size_t ws_size, hipStream_t stream) {
  const float* x      = (const float*)d_in[0];
  const float* w_qkv  = (const float*)d_in[1];
  const float* w_out  = (const float*)d_in[2];
  const float* b_out  = (const float*)d_in[3];
  const float* ggamma = (const float*)d_in[4];
  const float* gbeta  = (const float*)d_in[5];
  float* out = (float*)d_out;

  // Workspace carve-up (~24.2 MB)
  const size_t NBH = 4 * NHEAD;                    // 16 (b,head) pairs
  __bf16* Qb = (__bf16*)d_ws;                      // [bh][n][d]  4 MB
  __bf16* Kb = Qb + NBH * NPIX * DH;               // [bh][n][d]  4 MB
  __bf16* Vt = Kb + NBH * NPIX * DH;               // [bh][d][n]  4 MB
  __bf16* O  = Vt + NBH * NPIX * DH;               // [b][n][c]   4 MB
  float*  y  = (float*)(O + (size_t)4 * NPIX * CCH);  // [b][c][n] 8 MB
  float* gsum = y + (size_t)4 * CCH * NPIX;        // 16 floats
  float* gss  = gsum + 16;                         // 16 floats
  __bf16* wqkv_bf = (__bf16*)(gss + 16);           // 96 KB
  __bf16* wout_bf = wqkv_bf + (size_t)3 * CCH * CCH;  // 32 KB

  prep_kernel<<<192, 256, 0, stream>>>(w_qkv, w_out, wqkv_bf, wout_bf, gsum);
  qkv_kernel<<<dim3(32, 4), 256, 0, stream>>>(x, wqkv_bf, Qb, Kb, Vt);
  attn_kernel<<<dim3(32, 16), 256, 0, stream>>>(Qb, Kb, Vt, O);
  outproj_kernel<<<dim3(32, 4), 256, 0, stream>>>(O, wout_bf, b_out, y, gsum, gss);
  gn_kernel<<<2048, 256, 0, stream>>>(y, x, ggamma, gbeta, gsum, gss, out);
}
